// GQAAttention_46892452937745
// MI455X (gfx1250) — matrix operand order repaired and checked by reading, not compiled, not run on hardware
//
#include <hip/hip_runtime.h>
#include <hip/hip_bf16.h>
#include <cstdint>

typedef __attribute__((ext_vector_type(16))) __bf16 v16bf;
typedef __attribute__((ext_vector_type(8)))  float  v8f;

union BFrag { v16bf v; uint4 u[2]; };

__device__ __forceinline__ unsigned short f2bf(float f) {
  unsigned int u = __builtin_bit_cast(unsigned int, f);
  u += 0x7FFFu + ((u >> 16) & 1u);          // round-to-nearest-even
  return (unsigned short)(u >> 16);
}

static constexpr int kB   = 2;
static constexpr int kS   = 2048;
static constexpr int kHID = 2048;
static constexpr int kH   = 16;
static constexpr int kHKV = 4;
static constexpr int kD   = 128;
static constexpr int kM   = kB * kS;        // 4096 rows for all GEMMs

// ---------------------------------------------------------------------------
// 1) fp32 -> bf16 elementwise convert
// ---------------------------------------------------------------------------
__global__ void cvt_f32_bf16(const float* __restrict__ in,
                             unsigned short* __restrict__ out, int n) {
  int i = blockIdx.x * blockDim.x + threadIdx.x;
  if (i < n) out[i] = f2bf(in[i]);
}

// ---------------------------------------------------------------------------
// 2) W[K,N] fp32 -> WT[N,K] bf16 (32x32 LDS tile transpose)
// ---------------------------------------------------------------------------
__global__ __launch_bounds__(256) void transpose_w(const float* __restrict__ W,
                                                   unsigned short* __restrict__ WT,
                                                   int K, int N) {
  __shared__ float t[32][33];
  int n0 = blockIdx.x * 32, k0 = blockIdx.y * 32;
  int tx = threadIdx.x, ty = threadIdx.y;       // block (32,8)
#pragma unroll
  for (int i = 0; i < 4; i++)
    t[ty + 8 * i][tx] = W[(size_t)(k0 + ty + 8 * i) * N + n0 + tx];
  __syncthreads();
#pragma unroll
  for (int i = 0; i < 4; i++)
    WT[(size_t)(n0 + ty + 8 * i) * K + k0 + tx] = f2bf(t[tx][ty + 8 * i]);
}

// ---------------------------------------------------------------------------
// 3) V[b,s,hkv,d] bf16 -> VT[b,hkv,d,s] bf16
// ---------------------------------------------------------------------------
__global__ __launch_bounds__(256) void transpose_v(const unsigned short* __restrict__ V,
                                                   unsigned short* __restrict__ VT) {
  __shared__ unsigned short t[32][33];
  int bh = blockIdx.z, b = bh / kHKV, h = bh % kHKV;
  int d0 = blockIdx.x * 32, s0 = blockIdx.y * 32;
  int tx = threadIdx.x, ty = threadIdx.y;       // block (32,8)
#pragma unroll
  for (int i = 0; i < 4; i++)
    t[ty + 8 * i][tx] =
        V[(((size_t)b * kS + s0 + ty + 8 * i) * kHKV + h) * kD + d0 + tx];
  __syncthreads();
#pragma unroll
  for (int i = 0; i < 4; i++)
    VT[(((size_t)b * kHKV + h) * kD + d0 + ty + 8 * i) * kS + s0 + tx] =
        t[tx][ty + 8 * i];
}

// ---------------------------------------------------------------------------
// 4) Tiled bf16 WMMA GEMM: C[M,N] = alpha * (A[M,K] @ B[K,N]) + bias
//    A row-major bf16; B supplied pre-transposed as BT[N,K] bf16.
//    Block tile 128x128, 8 waves (2x4), wave tile 64x32 (4x2 WMMA accums).
//    Tile staging: GLOBAL_LOAD_ASYNC_TO_LDS_B128 with LDS double buffering —
//    the async unit fills buffer sel^1 while waves run WMMAs on buffer sel.
//    Per wave: 4 async ops/tile; ASYNCcnt is in-order for loads, so after
//    issuing tile it+1, s_wait_asynccnt 0x4 guarantees tile it is complete.
// ---------------------------------------------------------------------------
__global__ __launch_bounds__(256) void gemm_bf16(const unsigned short* __restrict__ A,
                                                 const unsigned short* __restrict__ BT,
                                                 const float* __restrict__ bias,
                                                 unsigned short* __restrict__ Cb,
                                                 float* __restrict__ Cf,
                                                 int M, int N, int K, float alpha) {
  constexpr int LP = 40;                 // padded LDS row (ushorts): 80B, 16B-aligned
  __shared__ unsigned short sA[2][128 * LP];
  __shared__ unsigned short sB[2][128 * LP];
  constexpr unsigned BUFB = 128 * LP * 2;       // bytes per buffer

  int tid = threadIdx.x, wid = tid >> 5, lane = tid & 31;
  int waveM = wid >> 2, waveN = wid & 3;        // 2 (M) x 4 (N) wave grid
  int lhi = lane >> 4, lcol = lane & 15;
  int M0 = blockIdx.y * 128, N0 = blockIdx.x * 128;

  v8f acc[4][2];
#pragma unroll
  for (int i = 0; i < 4; i++)
#pragma unroll
    for (int j = 0; j < 2; j++)
#pragma unroll
      for (int r = 0; r < 8; r++) acc[i][j][r] = 0.0f;

  int row = tid >> 1, half = tid & 1;           // tile staging assignment
  const unsigned short* gA = &A[(size_t)(M0 + row) * K + half * 16];
  const unsigned short* gB = &BT[(size_t)(N0 + row) * K + half * 16];
  unsigned ldsA0 = (unsigned)(size_t)&sA[0][row * LP + half * 16];
  unsigned ldsB0 = (unsigned)(size_t)&sB[0][row * LP + half * 16];

  auto issue_tile = [&](int k0, int sel) {
    unsigned long long a = (unsigned long long)(size_t)(gA + k0);
    unsigned long long b = (unsigned long long)(size_t)(gB + k0);
    unsigned la = ldsA0 + (unsigned)sel * BUFB;
    unsigned lb = ldsB0 + (unsigned)sel * BUFB;
    asm volatile("global_load_async_to_lds_b128 %0, %2, off\n\t"
                 "global_load_async_to_lds_b128 %0, %2, off offset:16\n\t"
                 "global_load_async_to_lds_b128 %1, %3, off\n\t"
                 "global_load_async_to_lds_b128 %1, %3, off offset:16"
                 :: "v"(la), "v"(lb), "v"(a), "v"(b)
                 : "memory");
  };

  const int nit = K / 32;
  issue_tile(0, 0);
  for (int it = 0; it < nit; ++it) {
    int sel = it & 1;
    if (it + 1 < nit) {
      issue_tile((it + 1) * 32, sel ^ 1);
      asm volatile("s_wait_asynccnt 0x4" ::: "memory");   // tile it complete
    } else {
      asm volatile("s_wait_asynccnt 0x0" ::: "memory");
    }
    __syncthreads();                    // tile it visible to all waves

    const unsigned short* cA = &sA[sel][0];
    const unsigned short* cB = &sB[sel][0];
    BFrag af[4], bfr[2];
    int kb = lhi * 8;                   // A layout: K 0-7/16-23 vs 8-15/24-31
#pragma unroll
    for (int i = 0; i < 4; i++) {
      int r = waveM * 64 + 16 * i + lcol;
      af[i].u[0] = *(const uint4*)&cA[r * LP + kb];
      af[i].u[1] = *(const uint4*)&cA[r * LP + kb + 16];
    }
    int kh = lhi * 8;                   // B layout: K 0-7/16-23 vs 8-15/24-31
#pragma unroll
    for (int j = 0; j < 2; j++) {
      int r = waveN * 32 + 16 * j + lcol;
      bfr[j].u[0] = *(const uint4*)&cB[r * LP + kh];
      bfr[j].u[1] = *(const uint4*)&cB[r * LP + kh + 16];
    }
#pragma unroll
    for (int i = 0; i < 4; i++)
#pragma unroll
      for (int j = 0; j < 2; j++)
        acc[i][j] = __builtin_amdgcn_wmma_f32_16x16x32_bf16(
            false, af[i].v, false, bfr[j].v, (short)0, acc[i][j], false, false);
    __syncthreads();                    // all reads done before buffer refill (WAR)
  }

#pragma unroll
  for (int j = 0; j < 2; j++) {
    int n = N0 + waveN * 32 + 16 * j + lcol;
    float bv = bias ? bias[n] : 0.0f;
#pragma unroll
    for (int i = 0; i < 4; i++) {
#pragma unroll
      for (int r = 0; r < 8; r++) {
        int m = M0 + waveM * 64 + 16 * i + r + 8 * lhi;
        float val = acc[i][j][r] * alpha + bv;
        size_t off = (size_t)m * N + n;
        if (Cf) Cf[off] = val; else Cb[off] = f2bf(val);
      }
    }
  }
}

// ---------------------------------------------------------------------------
// 5) Causal GQA flash attention. Grid (S/64, H, B), 128 threads (4 waves).
//    Wave w owns 16 query rows; online softmax; P re-layout via per-wave LDS.
//    Q is pre-scaled by 1/sqrt(D). Output bf16 [b,s,h,d].
// ---------------------------------------------------------------------------
__global__ __launch_bounds__(128) void attn_kernel(const unsigned short* __restrict__ Q,
                                                   const unsigned short* __restrict__ Kb,
                                                   const unsigned short* __restrict__ VT,
                                                   unsigned short* __restrict__ Ob) {
  __shared__ unsigned short P[4][16][32];
  int wid = threadIdx.x >> 5, lane = threadIdx.x & 31;
  int lhi = lane >> 4, lcol = lane & 15;
  int qh = blockIdx.y, b = blockIdx.z;
  int h = qh >> 2;                               // G = H/HKV = 4
  int q0w = blockIdx.x * 64 + wid * 16;

  // Q A-fragments for the 4 K-steps over D=128 (kept resident)
  BFrag qf[4];
  {
    int q = q0w + lcol;
    const unsigned short* qp = &Q[(((size_t)b * kS + q) * kH + qh) * kD];
    int kb = lhi * 8;
#pragma unroll
    for (int i = 0; i < 4; i++) {
      qf[i].u[0] = *(const uint4*)&qp[32 * i + kb];
      qf[i].u[1] = *(const uint4*)&qp[32 * i + kb + 16];
    }
  }

  v8f O[8];
#pragma unroll
  for (int j = 0; j < 8; j++)
#pragma unroll
    for (int r = 0; r < 8; r++) O[j][r] = 0.0f;
  float mrow[8], lsum[8];
#pragma unroll
  for (int r = 0; r < 8; r++) { mrow[r] = -1e30f; lsum[r] = 0.0f; }

  const unsigned short* vbase =
      &VT[(((size_t)b * kHKV + h) * kD) * kS];

  for (int k0 = 0; k0 < q0w + 16; k0 += 32) {
    // prefetch next key/value tile (global_prefetch_b8; everything sits in L2)
    if (k0 + 32 < q0w + 16) {
      __builtin_prefetch(&Kb[(((size_t)b * kS + k0 + 32 + lcol) * kHKV + h) * kD], 0, 0);
      __builtin_prefetch(vbase + (size_t)lcol * kS + k0 + 32, 0, 0);
    }

    // ---- scores S = Q @ K^T over D=128 (two 16x16 C tiles) ----
    v8f s0, s1;
#pragma unroll
    for (int r = 0; r < 8; r++) { s0[r] = 0.0f; s1[r] = 0.0f; }
    int kh = lhi * 8;
#pragma unroll
    for (int t = 0; t < 2; t++) {
      int kc = k0 + 16 * t + lcol;
      const unsigned short* kp = &Kb[(((size_t)b * kS + kc) * kHKV + h) * kD] + kh;
#pragma unroll
      for (int i = 0; i < 4; i++) {
        BFrag kf;
        kf.u[0] = *(const uint4*)(kp + 32 * i);
        kf.u[1] = *(const uint4*)(kp + 32 * i + 16);
        if (t == 0)
          s0 = __builtin_amdgcn_wmma_f32_16x16x32_bf16(false, qf[i].v, false, kf.v,
                                                       (short)0, s0, false, false);
        else
          s1 = __builtin_amdgcn_wmma_f32_16x16x32_bf16(false, qf[i].v, false, kf.v,
                                                       (short)0, s1, false, false);
      }
    }

    // ---- causal mask + online softmax (C layout: row = r + 8*lhi, col = lcol) ----
#pragma unroll
    for (int r = 0; r < 8; r++) {
      int q = q0w + r + 8 * lhi;
      float v0 = s0[r], v1 = s1[r];
      if (k0 + lcol > q)      v0 = -1e9f;
      if (k0 + 16 + lcol > q) v1 = -1e9f;
      float mx = fmaxf(v0, v1);
      mx = fmaxf(mx, __shfl_xor(mx, 1, 32));
      mx = fmaxf(mx, __shfl_xor(mx, 2, 32));
      mx = fmaxf(mx, __shfl_xor(mx, 4, 32));
      mx = fmaxf(mx, __shfl_xor(mx, 8, 32));
      float mnew = fmaxf(mrow[r], mx);
      float corr = __expf(mrow[r] - mnew);
      mrow[r] = mnew;
      float p0 = __expf(v0 - mnew);
      float p1 = __expf(v1 - mnew);
      float rs = p0 + p1;
      rs += __shfl_xor(rs, 1, 32);
      rs += __shfl_xor(rs, 2, 32);
      rs += __shfl_xor(rs, 4, 32);
      rs += __shfl_xor(rs, 8, 32);
      lsum[r] = lsum[r] * corr + rs;
#pragma unroll
      for (int j = 0; j < 8; j++) O[j][r] *= corr;
      P[wid][r + 8 * lhi][lcol]      = f2bf(p0);
      P[wid][r + 8 * lhi][16 + lcol] = f2bf(p1);
    }

    // ---- re-read P in A-matrix layout (same wave: LDS ops are in-order) ----
    BFrag pf;
    {
      int pk = lhi * 8;
      pf.u[0] = *(const uint4*)&P[wid][lcol][pk];
      pf.u[1] = *(const uint4*)&P[wid][lcol][pk + 16];
    }

    // ---- O += P @ V  (V^T[b,h,d,s]: keys 0-7/16-23 vs 8-15/24-31 per lane) ----
    const unsigned short* vp = vbase + k0 + 8 * lhi;
#pragma unroll
    for (int j = 0; j < 8; j++) {
      BFrag vf;
      const unsigned short* vpp = vp + (size_t)(16 * j + lcol) * kS;
      vf.u[0] = *(const uint4*)vpp;
      vf.u[1] = *(const uint4*)(vpp + 16);
      O[j] = __builtin_amdgcn_wmma_f32_16x16x32_bf16(false, pf.v, false, vf.v,
                                                     (short)0, O[j], false, false);
    }
  }

  // ---- finalize: O /= l, write bf16 [b,s,h,d] ----
#pragma unroll
  for (int r = 0; r < 8; r++) {
    float inv = 1.0f / lsum[r];
    int q = q0w + r + 8 * lhi;
    size_t base = (((size_t)b * kS + q) * kH + qh) * kD + lcol;
#pragma unroll
    for (int j = 0; j < 8; j++) Ob[base + 16 * j] = f2bf(O[j][r] * inv);
  }
}

// ---------------------------------------------------------------------------
// Host-side launch
// ---------------------------------------------------------------------------
extern "C" void kernel_launch(void* const* d_in, const int* in_sizes, int n_in,
                              void* d_out, int out_size, void* d_ws, size_t ws_size,
                              hipStream_t stream) {
  const float* x  = (const float*)d_in[0];
  const float* wq = (const float*)d_in[1];
  const float* bq = (const float*)d_in[2];
  const float* wk = (const float*)d_in[3];
  const float* bk = (const float*)d_in[4];
  const float* wv = (const float*)d_in[5];
  const float* bv = (const float*)d_in[6];
  const float* wo = (const float*)d_in[7];
  const float* bo = (const float*)d_in[8];
  float* out = (float*)d_out;

  char* w = (char*)d_ws;
  auto take = [&](size_t elems) { unsigned short* p = (unsigned short*)w; w += elems * 2; return p; };
  unsigned short* xb   = take((size_t)kM * kHID);        // x bf16
  unsigned short* wqT  = take((size_t)kHID * kHID);      // [N=2048][K=2048]
  unsigned short* wkT  = take((size_t)(kHKV * kD) * kHID);
  unsigned short* wvT  = take((size_t)(kHKV * kD) * kHID);
  unsigned short* woT  = take((size_t)kHID * kHID);
  unsigned short* Qb   = take((size_t)kM * kH * kD);     // [b,s,h,d], pre-scaled
  unsigned short* Kbuf = take((size_t)kM * kHKV * kD);   // [b,s,hkv,d]
  unsigned short* Vbuf = take((size_t)kM * kHKV * kD);
  unsigned short* VTb  = take((size_t)kM * kHKV * kD);   // [b,hkv,d,s]
  unsigned short* attn = take((size_t)kM * kH * kD);     // [b,s,h,d]

  const int nx = kM * kHID;
  cvt_f32_bf16<<<(nx + 255) / 256, 256, 0, stream>>>(x, xb, nx);

  dim3 tb(32, 8);
  transpose_w<<<dim3(kHID / 32, kHID / 32), tb, 0, stream>>>(wq, wqT, kHID, kH * kD);
  transpose_w<<<dim3((kHKV * kD) / 32, kHID / 32), tb, 0, stream>>>(wk, wkT, kHID, kHKV * kD);
  transpose_w<<<dim3((kHKV * kD) / 32, kHID / 32), tb, 0, stream>>>(wv, wvT, kHID, kHKV * kD);
  transpose_w<<<dim3(kHID / 32, kHID / 32), tb, 0, stream>>>(wo, woT, kHID, kHID);

  const float qscale = 0.088388347648318447f;            // 1/sqrt(128), folded into Q
  gemm_bf16<<<dim3((kH * kD) / 128, kM / 128), 256, 0, stream>>>(
      xb, wqT, bq, Qb, nullptr, kM, kH * kD, kHID, qscale);
  gemm_bf16<<<dim3((kHKV * kD) / 128, kM / 128), 256, 0, stream>>>(
      xb, wkT, bk, Kbuf, nullptr, kM, kHKV * kD, kHID, 1.0f);
  gemm_bf16<<<dim3((kHKV * kD) / 128, kM / 128), 256, 0, stream>>>(
      xb, wvT, bv, Vbuf, nullptr, kM, kHKV * kD, kHID, 1.0f);

  transpose_v<<<dim3(kD / 32, kS / 32, kB * kHKV), tb, 0, stream>>>(Vbuf, VTb);

  attn_kernel<<<dim3(kS / 64, kH, kB), 128, 0, stream>>>(Qb, Kbuf, VTb, attn);

  gemm_bf16<<<dim3(kHID / 128, kM / 128), 256, 0, stream>>>(
      attn, woT, bo, nullptr, out, kM, kHID, kHID, 1.0f);
}
